// RNN_15556371546612
// MI455X (gfx1250) — matrix operand order repaired and checked by reading, not compiled, not run on hardware
//
#include <hip/hip_runtime.h>
#include <hip/hip_bf16.h>

// Problem constants (match reference)
#define BB 32
#define TT 4096
#define DD 512
#define UU 512

typedef __bf16 bf16;
typedef __attribute__((ext_vector_type(16))) __bf16 v16bf;
typedef __attribute__((ext_vector_type(8)))  __bf16 v8bf;
typedef __attribute__((ext_vector_type(8)))  float  v8f;
typedef __attribute__((ext_vector_type(4)))  float  v4f;

__device__ __forceinline__ v8f wmma_bf16(v16bf a, v16bf b, v8f c) {
  return __builtin_amdgcn_wmma_f32_16x16x32_bf16(false, a, false, b, (short)0, c, false, false);
}

// ---------------------------------------------------------------------------
// Kernel 0: zero the inter-workgroup barrier counters (fresh state every call)
// ---------------------------------------------------------------------------
__global__ void init_cnt_kernel(unsigned* __restrict__ cnt) {
  if (threadIdx.x < 2) cnt[threadIdx.x] = 0u;
}

// ---------------------------------------------------------------------------
// Kernel 1: XW = x @ w_x + b_h   (M = B*T = 131072, N = 512, K = 512)
// bf16 WMMA, LDS-staged tiles. Block tile 128(M) x 128(N), K-step 32.
// 256 threads = 8 waves; wave w owns N-tile w and iterates all 8 M-tiles.
// ---------------------------------------------------------------------------
__global__ __launch_bounds__(256) void xw_gemm_kernel(
    const float* __restrict__ x, const float* __restrict__ w_x,
    const float* __restrict__ b_h, float* __restrict__ xw)
{
  constexpr int BM = 128, BN = 128, BK = 32;
  __shared__ bf16 As[BM][BK];   // row-major [m][k]   8 KB
  __shared__ bf16 Bs[BN][BK];   // col-major [n][k]   8 KB

  const int tid  = threadIdx.x;
  const int m0   = blockIdx.y * BM;
  const int n0   = blockIdx.x * BN;
  const int wave = tid >> 5;    // N-tile 0..7
  const int lane = tid & 31;
  const int half = lane >> 4;
  const int l16  = lane & 15;

  const float bias = b_h[n0 + wave * 16 + l16];
  v8f acc[8];
  for (int i = 0; i < 8; ++i)
    for (int r = 0; r < 8; ++r) acc[i][r] = bias;

  for (int k0 = 0; k0 < DD; k0 += BK) {
    __syncthreads();
    // stage A: 128x32 fp32 -> bf16 (4 x float4 per thread, coalesced)
    for (int i = 0; i < 4; ++i) {
      int idx = (tid + i * 256) * 4;            // 0..4095
      int am = idx >> 5, ak = idx & 31;
      v4f v = *(const v4f*)&x[(size_t)(m0 + am) * DD + k0 + ak];
      As[am][ak + 0] = (bf16)v.x; As[am][ak + 1] = (bf16)v.y;
      As[am][ak + 2] = (bf16)v.z; As[am][ak + 3] = (bf16)v.w;
    }
    // stage B: 32x128 fp32 -> bf16, transposed to column-major
    for (int i = 0; i < 4; ++i) {
      int idx = (tid + i * 256) * 4;            // 0..4095
      int bk = idx >> 7, bn = idx & 127;
      v4f v = *(const v4f*)&w_x[(size_t)(k0 + bk) * UU + n0 + bn];
      Bs[bn + 0][bk] = (bf16)v.x; Bs[bn + 1][bk] = (bf16)v.y;
      Bs[bn + 2][bk] = (bf16)v.z; Bs[bn + 3][bk] = (bf16)v.w;
    }
    __syncthreads();

    // B fragment: lane -> column n = wave*16+l16, K = {half*8..+7, 16+half*8..+7}
    union { v16bf v; v8bf h[2]; } ub;
    ub.h[0] = *(const v8bf*)&Bs[wave * 16 + l16][half * 8];
    ub.h[1] = *(const v8bf*)&Bs[wave * 16 + l16][16 + half * 8];

    for (int i = 0; i < 8; ++i) {
      // A fragment: lane -> row l16, K = {half*8..+7, 16+half*8..+7}
      union { v16bf v; v8bf h[2]; } ua;
      ua.h[0] = *(const v8bf*)&As[i * 16 + l16][half * 8];
      ua.h[1] = *(const v8bf*)&As[i * 16 + l16][16 + half * 8];
      acc[i] = wmma_bf16(ua.v, ub.v, acc[i]);
    }
  }

  // store fp32 result (C layout: VGPR r -> M = r + half*8, N = l16)
  const int gn = n0 + wave * 16 + l16;
  for (int i = 0; i < 8; ++i) {
    for (int r = 0; r < 8; ++r) {
      int gm = m0 + i * 16 + r + half * 8;
      xw[(size_t)gm * UU + gn] = acc[i][r];
    }
  }
}

// ---------------------------------------------------------------------------
// Kernel 2: persistent recurrent kernel (software-pipelined).
// 16 WGs: blockIdx = g*8 + s ; g = batch group (16 rows), s = N slice (64 cols).
// Weight slices live in LDS (bf16, column-major). Full state s_t (16x512 bf16)
// is exchanged through L2 each step (async-to-LDS pull) with a monotonic
// atomic barrier per group. 128 threads = 4 waves; wave w owns N-tile w.
//
// Retimed recurrence: at iteration t, hs holds s_t. A single fused k-loop
// computes BOTH  g_t = s_t @ w_h  and  y_{t-1} = s_t @ w_y  as 4 independent
// WMMA chains sharing A-fragments; then s_{t+1} = tanh(g_t + xw_t) is
// published/exchanged. y stores overlap the barrier wait.
// ---------------------------------------------------------------------------
__global__ __launch_bounds__(128) void rnn_persistent_kernel(
    const float* __restrict__ w_h, const float* __restrict__ w_y,
    const float* __restrict__ b_y, const float* __restrict__ h0,
    const float* __restrict__ xw,
    float* __restrict__ ys, float* __restrict__ h_final,
    bf16* __restrict__ hbuf,          // [2 groups][2 phases][16][512] bf16
    unsigned* __restrict__ cnt)       // [2] barrier counters
{
  constexpr int NS = 64;   // N slice width
  constexpr int GB = 16;   // batch rows per group
  __shared__ bf16 whs[NS][DD];   // w_h slice, col-major  64 KB
  __shared__ bf16 wys[NS][DD];   // w_y slice, col-major  64 KB
  __shared__ bf16 hs[GB][DD];    // full state s_t (bf16) 16 KB
  __shared__ float bys[NS];

  const int tid  = threadIdx.x;
  const int wave = tid >> 5;       // N-tile 0..3
  const int lane = tid & 31;
  const int half = lane >> 4;
  const int l16  = lane & 15;
  const int g    = blockIdx.x >> 3;
  const int s    = blockIdx.x & 7;
  const int n0   = s * NS;
  const int gn   = n0 + wave * 16 + l16;   // global output column of this lane

  // LDS byte offset of hs (flat shared address truncates to LDS offset)
  const unsigned hs_lds = (unsigned)(size_t)&hs[0][0];

  // one-time: weight slices -> LDS (col-major, bf16), biases, h0 broadcast
  for (int idx = tid; idx < NS * DD; idx += 128) {
    int k = idx >> 6, c = idx & 63;
    whs[c][k] = (bf16)w_h[(size_t)k * UU + n0 + c];
    wys[c][k] = (bf16)w_y[(size_t)k * UU + n0 + c];
  }
  for (int c = tid; c < NS; c += 128) bys[c] = b_y[n0 + c];
  for (int c = tid; c < DD; c += 128) {
    bf16 v = (bf16)h0[c];
    for (int m = 0; m < GB; ++m) hs[m][c] = v;
  }
  __syncthreads();

  const float by = bys[wave * 16 + l16];

  for (int t = 0; t < TT; ++t) {
    // issue xw loads for this step early; consumed only after the WMMA chains
    float xr[8];
    for (int r = 0; r < 8; ++r) {
      int m = r + half * 8;
      xr[r] = xw[((size_t)(g * GB + m) * TT + t) * UU + gn];
    }
    if (t + 1 < TT) {  // prefetch next step's xw row (global_prefetch_b8)
      __builtin_prefetch(&xw[((size_t)(g * GB + half * 8) * TT + (t + 1)) * UU + gn], 0, 1);
    }

    // ---- fused GEMMs on s_t: 4 independent WMMA chains, shared A frags ----
    v8f ha0, ha1, ya0, ya1;
    for (int r = 0; r < 8; ++r) { ha0[r] = 0.f; ha1[r] = 0.f; ya0[r] = 0.f; ya1[r] = 0.f; }
    for (int k0 = 0; k0 < DD; k0 += 64) {
      union { v16bf v; v8bf h[2]; } ua0, ua1, ubh0, ubh1, uby0, uby1;
      ua0.h[0] = *(const v8bf*)&hs[l16][k0 + half * 8];
      ua0.h[1] = *(const v8bf*)&hs[l16][k0 + 16 + half * 8];
      ua1.h[0] = *(const v8bf*)&hs[l16][k0 + 32 + half * 8];
      ua1.h[1] = *(const v8bf*)&hs[l16][k0 + 48 + half * 8];
      ubh0.h[0] = *(const v8bf*)&whs[wave * 16 + l16][k0 + half * 8];
      ubh0.h[1] = *(const v8bf*)&whs[wave * 16 + l16][k0 + 16 + half * 8];
      ubh1.h[0] = *(const v8bf*)&whs[wave * 16 + l16][k0 + 32 + half * 8];
      ubh1.h[1] = *(const v8bf*)&whs[wave * 16 + l16][k0 + 48 + half * 8];
      uby0.h[0] = *(const v8bf*)&wys[wave * 16 + l16][k0 + half * 8];
      uby0.h[1] = *(const v8bf*)&wys[wave * 16 + l16][k0 + 16 + half * 8];
      uby1.h[0] = *(const v8bf*)&wys[wave * 16 + l16][k0 + 32 + half * 8];
      uby1.h[1] = *(const v8bf*)&wys[wave * 16 + l16][k0 + 48 + half * 8];
      ha0 = wmma_bf16(ua0.v, ubh0.v, ha0);
      ya0 = wmma_bf16(ua0.v, uby0.v, ya0);
      ha1 = wmma_bf16(ua1.v, ubh1.v, ha1);
      ya1 = wmma_bf16(ua1.v, uby1.v, ya1);
    }

    // y_{t-1} = s_t @ w_y + b_y  (discard at t==0); overlaps the barrier wait
    if (t > 0) {
      for (int r = 0; r < 8; ++r) {
        int m = r + half * 8;
        ys[((size_t)(g * GB + m) * TT + (t - 1)) * UU + gn] = ya0[r] + ya1[r] + by;
      }
    }

    // s_{t+1} = tanh(g_t + xw_t)
    float hv[8];
    for (int r = 0; r < 8; ++r) hv[r] = tanhf(ha0[r] + ha1[r] + xr[r]);

    // publish bf16 slice (double-buffered by t parity)
    bf16* hb = hbuf + ((size_t)g * 2 + (t & 1)) * (GB * UU);
    for (int r = 0; r < 8; ++r) {
      int m = r + half * 8;
      hb[m * UU + gn] = (bf16)hv[r];
    }
    if (t == TT - 1) {  // h_final = s_T (fp32, exact)
      for (int r = 0; r < 8; ++r) {
        int m = r + half * 8;
        h_final[(size_t)(g * GB + m) * UU + gn] = hv[r];
      }
    }

    // ---- barrier across the 8 slice-WGs of this group (monotonic) ----
    __threadfence();
    __syncthreads();
    if (tid == 0) {
      atomicAdd(&cnt[g], 1u);
      unsigned target = 8u * (unsigned)(t + 1);
      while (__atomic_load_n(&cnt[g], __ATOMIC_RELAXED) < target) { }
    }
    __syncthreads();
    __threadfence();

    // ---- pull full s_{t+1} into LDS via async copy (ASYNCcnt path) ----
    {
      const char* src = (const char*)(hbuf + ((size_t)g * 2 + (t & 1)) * (GB * UU));
      for (int i = 0; i < 8; ++i) {
        unsigned off = (unsigned)(tid + i * 128) * 16u;   // 16 KB total
        asm volatile("global_load_async_to_lds_b128 %0, %1, off"
                     :: "v"(hs_lds + off), "v"(src + off) : "memory");
      }
      asm volatile("s_wait_asynccnt 0x0" ::: "memory");
    }
    __syncthreads();
  }

  // ---- epilogue: y_{T-1} = s_T @ w_y + b_y ----
  {
    v8f ya0, ya1;
    for (int r = 0; r < 8; ++r) { ya0[r] = 0.f; ya1[r] = 0.f; }
    for (int k0 = 0; k0 < DD; k0 += 64) {
      union { v16bf v; v8bf h[2]; } ua, ub;
      ua.h[0] = *(const v8bf*)&hs[l16][k0 + half * 8];
      ua.h[1] = *(const v8bf*)&hs[l16][k0 + 16 + half * 8];
      ub.h[0] = *(const v8bf*)&wys[wave * 16 + l16][k0 + half * 8];
      ub.h[1] = *(const v8bf*)&wys[wave * 16 + l16][k0 + 16 + half * 8];
      ya0 = wmma_bf16(ua.v, ub.v, ya0);
      ua.h[0] = *(const v8bf*)&hs[l16][k0 + 32 + half * 8];
      ua.h[1] = *(const v8bf*)&hs[l16][k0 + 48 + half * 8];
      ub.h[0] = *(const v8bf*)&wys[wave * 16 + l16][k0 + 32 + half * 8];
      ub.h[1] = *(const v8bf*)&wys[wave * 16 + l16][k0 + 48 + half * 8];
      ya1 = wmma_bf16(ua.v, ub.v, ya1);
    }
    for (int r = 0; r < 8; ++r) {
      int m = r + half * 8;
      ys[((size_t)(g * GB + m) * TT + (TT - 1)) * UU + gn] = ya0[r] + ya1[r] + by;
    }
  }
}

// ---------------------------------------------------------------------------
extern "C" void kernel_launch(void* const* d_in, const int* in_sizes, int n_in,
                              void* d_out, int out_size, void* d_ws, size_t ws_size,
                              hipStream_t stream) {
  const float* x   = (const float*)d_in[0];
  const float* w_h = (const float*)d_in[1];
  const float* w_x = (const float*)d_in[2];
  const float* w_y = (const float*)d_in[3];
  const float* b_h = (const float*)d_in[4];
  const float* b_y = (const float*)d_in[5];
  const float* h0  = (const float*)d_in[6];

  float* out     = (float*)d_out;
  float* ys      = out;                                   // [B][T][U]
  float* h_final = out + (size_t)BB * TT * UU;            // [B][U]

  // workspace layout: xw fp32 [B*T][U] | hbuf bf16 [2][2][16][512] | cnt[2]
  float*    xw   = (float*)d_ws;
  size_t    xwB  = (size_t)BB * TT * UU * sizeof(float);
  bf16*     hbuf = (bf16*)((char*)d_ws + xwB);
  size_t    hbB  = (size_t)2 * 2 * 16 * UU * sizeof(bf16);
  unsigned* cnt  = (unsigned*)((char*)d_ws + xwB + hbB);

  hipLaunchKernelGGL(init_cnt_kernel, dim3(1), dim3(32), 0, stream, cnt);
  hipLaunchKernelGGL(xw_gemm_kernel, dim3(UU / 128, (BB * TT) / 128), dim3(256),
                     0, stream, x, w_x, b_h, xw);
  hipLaunchKernelGGL(rnn_persistent_kernel, dim3(16), dim3(128), 0, stream,
                     w_h, w_y, b_y, h0, xw, ys, h_final, hbuf, cnt);
}
